// rnn_cov_36558761623655
// MI455X (gfx1250) — matrix operand order repaired and checked by reading, not compiled, not run on hardware
//
#include <hip/hip_runtime.h>
#include <hip/hip_bf16.h>

typedef _Float16 f16;
typedef __attribute__((ext_vector_type(16))) _Float16 v16h;
typedef __attribute__((ext_vector_type(8)))  _Float16 v8h;
typedef __attribute__((ext_vector_type(8)))  float    v8f;
typedef unsigned int u32;
typedef __attribute__((ext_vector_type(4))) u32 u32x4;
typedef __attribute__((ext_vector_type(8))) int i32x8;
typedef __attribute__((ext_vector_type(4))) int i32x4;

#define B_N   16384
#define T_N   10
#define I_N   184
#define H_N   128
#define G_N   512           // 4*H
#define K1PAD 192           // I padded to multiple of 32
#define K2PAD 256           // 2*H
#define BT    (B_N * T_N)
#define BTILES (B_N / 16)   // 1024

// ---------------------------------------------------------------- helpers ---

__device__ __forceinline__ v8f wmma_f16(v16h a, v16h b, v8f c) {
    return __builtin_amdgcn_wmma_f32_16x16x32_f16(false, a, false, b,
                                                  (short)0, c, false, false);
}

// A-matrix fragment (16xK tile, row-major source, lda in halves).
// Lanes 0-15: row=lane, K {kb..kb+7, kb+16..kb+23};
// lanes 16-31: row=lane-16, K {kb+8..kb+15, kb+24..kb+31}.
__device__ __forceinline__ v16h load_a_frag(const f16* base, int lda,
                                            int m0, int kb, int lane) {
    int row = m0 + (lane & 15);
    int hi  = (lane >> 4) * 8;
    const f16* p = base + (size_t)row * lda + kb + hi;
    v8h lo = *(const v8h*)p;
    v8h hv = *(const v8h*)(p + 16);
    v16h a;
#pragma unroll
    for (int i = 0; i < 8; ++i) { a[i] = lo[i]; a[8 + i] = hv[i]; }
    return a;
}

// B-matrix fragment for D = A*B with B(k,n) = W[n][k] (row-major W, ldb halves).
__device__ __forceinline__ v16h load_b_frag(const f16* w, int ldb,
                                            int n0, int kb, int lane) {
    int n = n0 + (lane & 15);
    int k = kb + (lane >> 4) * 8;
    const f16* p = w + (size_t)n * ldb + k;
    v8h lo = *(const v8h*)p;
    v8h hv = *(const v8h*)(p + 16);
    v16h b;
#pragma unroll
    for (int i = 0; i < 8; ++i) { b[i] = lo[i]; b[8 + i] = hv[i]; }
    return b;
}

__device__ __forceinline__ float sigm_f(float x) {
    return 1.0f / (1.0f + __expf(-x));
}
__device__ __forceinline__ float tanh_f(float x) {
    float t = __expf(-2.0f * fabsf(x));
    float r = (1.0f - t) / (1.0f + t);
    return copysignf(r, x);
}

// ------------------------------------------------------------ cvt kernels ---

// f32 [rows x ks] -> f16 [rows x kd], zero-padded K tail.
__global__ void cvt_pad_f16(const float* __restrict__ src, f16* __restrict__ dst,
                            int rows, int ks, int kd) {
    size_t i = (size_t)blockIdx.x * blockDim.x + threadIdx.x;
    if (i >= (size_t)rows * kd) return;
    int r = (int)(i / kd), c = (int)(i % kd);
    dst[i] = (c < ks) ? (f16)src[(size_t)r * ks + c] : (f16)0.0f;
}

// x (B,T,I) f32 -> time-major padded f16: dst[(t*B + b)*K1PAD + c]
__global__ void cvt_x_tmajor(const float* __restrict__ src, f16* __restrict__ dst) {
    size_t i = (size_t)blockIdx.x * blockDim.x + threadIdx.x;
    if (i >= (size_t)BT * K1PAD) return;
    int c = (int)(i % K1PAD);
    size_t m = i / K1PAD;
    int t = (int)(m / B_N), b = (int)(m % B_N);
    dst[i] = (c < I_N) ? (f16)src[((size_t)b * T_N + t) * I_N + c] : (f16)0.0f;
}

// ---------------------------------------------------- xproj GEMM (f16 WMMA) --
// Time-major M (m' = t*B + b).  Output tile-transposed:
//   out[(mtile*1024 + dir*512 + n)*16 + row]   (f16)
// grid: (Mtiles/16, 8, 2); block 256 (8 waves); wave -> 2 M-tiles x 64 cols.
// B tile (64 x Kpad) staged in LDS once per block via the Tensor Data Mover.
__global__ __launch_bounds__(256)
void xproj_gemm(const f16* __restrict__ A, int lda, int kch,
                const f16* __restrict__ w0, const f16* __restrict__ w1,
                const float* __restrict__ bi0, const float* __restrict__ bh0,
                const float* __restrict__ bi1, const float* __restrict__ bh1,
                f16* __restrict__ out) {
    __shared__ __align__(16) f16 b_lds[64 * K2PAD];   // up to 32 KB

    int wave = threadIdx.x >> 5, lane = threadIdx.x & 31;
    int mt0  = blockIdx.x * 16 + wave * 2;            // first of 2 M-tiles
    int m0   = mt0 * 16;
    int n0   = blockIdx.y * 64;
    int dir  = blockIdx.z;
    const f16* w = dir ? w1 : w0;
    const float* bi = dir ? bi1 : bi0;
    const float* bh = dir ? bh1 : bh0;

    // ---- stage 64 x Kpad weight tile into LDS via TDM ----
    const f16* wsrc = w + (size_t)n0 * lda;
#if __has_builtin(__builtin_amdgcn_tensor_load_to_lds)
    if (threadIdx.x == 0) {
        u32 lds_off = (u32)(uintptr_t)(void*)b_lds;   // low 32 bits = LDS offset
        unsigned long long ga = (unsigned long long)(uintptr_t)wsrc;
        u32x4 g0;
        g0[0] = 1u;                                   // count=1 valid descriptor
        g0[1] = lds_off;                              // lds_addr
        g0[2] = (u32)ga;                              // global_addr[31:0]
        g0[3] = (u32)((ga >> 32) & 0x1FFFFFFu) | (2u << 30);  // addr[56:32] | type=2
        i32x8 g1;
        g1[0] = (int)(1u << 16);                      // data_size=1 (2 bytes)
        g1[1] = (int)((u32)(lda & 0xFFFF) << 16);     // tensor_dim0[15:0]
        g1[2] = (int)((u32)64 << 16);                 // tensor_dim0 hi | tensor_dim1=64
        g1[3] = (int)((u32)(lda & 0xFFFF) << 16);     // tensor_dim1 hi | tile_dim0=lda
        g1[4] = 64;                                   // tile_dim1=64, tile_dim2=0
        g1[5] = (int)lda;                             // tensor_dim0_stride low32
        g1[6] = 0;                                    // stride0 hi | stride1 lo
        g1[7] = 0;
        i32x4 z4 = {0, 0, 0, 0};
        i32x8 z8 = {0, 0, 0, 0, 0, 0, 0, 0};
        __builtin_amdgcn_tensor_load_to_lds(g0, g1, z4, z4, z8, 0);
    }
    __builtin_amdgcn_s_wait_tensorcnt(0);
#else
    for (int i = threadIdx.x * 8; i < 64 * lda; i += 256 * 8)
        *(v8h*)&b_lds[i] = *(const v8h*)&wsrc[i];
#endif
    __syncthreads();

    v8f acc[2][4];
#pragma unroll
    for (int s = 0; s < 2; ++s)
#pragma unroll
        for (int nt = 0; nt < 4; ++nt) acc[s][nt] = v8f{};

    // double-buffered A fragments (2 sub-tiles), B from LDS
    v16h va0 = load_a_frag(A, lda, m0,      0, lane);
    v16h va1 = load_a_frag(A, lda, m0 + 16, 0, lane);
    for (int kc = 0; kc < kch; ++kc) {
        int kb = kc * 32;
        v16h na0, na1;
        if (kc + 1 < kch) {
            na0 = load_a_frag(A, lda, m0,      kb + 32, lane);
            na1 = load_a_frag(A, lda, m0 + 16, kb + 32, lane);
        }
#pragma unroll
        for (int nt = 0; nt < 4; ++nt) {
            int n = nt * 16 + (lane & 15);
            int k = kb + (lane >> 4) * 8;
            const f16* p = &b_lds[n * lda + k];
            v8h lo = *(const v8h*)p;
            v8h hv = *(const v8h*)(p + 16);
            v16h vb;
#pragma unroll
            for (int i = 0; i < 8; ++i) { vb[i] = lo[i]; vb[8 + i] = hv[i]; }
            acc[0][nt] = wmma_f16(va0, vb, acc[0][nt]);
            acc[1][nt] = wmma_f16(va1, vb, acc[1][nt]);
        }
        va0 = na0; va1 = na1;
    }

    // epilogue: lane col = lane&15, rows = (lane>>4)*8 + r; tile-transposed store
    int col = lane & 15, rowb = (lane >> 4) * 8;
#pragma unroll
    for (int s = 0; s < 2; ++s) {
        size_t mtile = (size_t)(mt0 + s);
#pragma unroll
        for (int nt = 0; nt < 4; ++nt) {
            int gn = n0 + nt * 16 + col;              // 0..511 gate column
            float bsum = bi[gn] + bh[gn];
            v8h hs;
#pragma unroll
            for (int r = 0; r < 8; ++r) hs[r] = (f16)(acc[s][nt][r] + bsum);
            *(v8h*)(out + (mtile * 1024 + dir * 512 + gn) * 16 + rowb) = hs;
        }
    }
}

// --------------------------------------------------------- LSTM recurrence --
// grid: (B/16, 2 dirs); block 256 (8 waves). Wave w owns cols [16w,16w+16)
// of h/c and of each gate. W_hh fragments live in registers for all 10 steps.
// xp is tile-transposed, time-major: xp[((t*BTILES + bx)*1024 + col)*16 + row].
__global__ __launch_bounds__(256)
void lstm_layer(const f16* __restrict__ xp,
                const f16* __restrict__ whh0, const f16* __restrict__ whh1,
                f16* __restrict__ h_out,      // layer1: [t*B + b][256]
                f16* __restrict__ pooled,     // layer2: [b][512]
                int layer2) {
    __shared__ __align__(16) f16 h_lds[16 * 128];

    int wave = threadIdx.x >> 5, lane = threadIdx.x & 31;
    int bx = blockIdx.x, b0 = bx * 16;
    int dir = blockIdx.y;
    const f16* whh = dir ? whh1 : whh0;

    // resident B fragments: 4 gates x 4 K-chunks (W_hh is [512][128] f16)
    v16h bf[4][4];
#pragma unroll
    for (int g = 0; g < 4; ++g)
#pragma unroll
        for (int kc = 0; kc < 4; ++kc)
            bf[g][kc] = load_b_frag(whh, 128, 128 * g + 16 * wave, kc * 32, lane);

    float cst[8], pm0[8], pm1[8];
#pragma unroll
    for (int r = 0; r < 8; ++r) { cst[r] = 0.f; pm0[r] = -3.0e38f; pm1[r] = -3.0e38f; }

    for (int i = threadIdx.x; i < 16 * 128; i += 256) h_lds[i] = (f16)0.0f;
    __syncthreads();

    int col = lane & 15, rowb = (lane >> 4) * 8;

    for (int tt = 0; tt < T_N; ++tt) {
        int t = dir ? (T_N - 1 - tt) : tt;
        size_t ctile = ((size_t)t * BTILES + bx) * 1024 + dir * 512;

        if (tt + 1 < T_N) {     // prefetch next step's gate tile
            int tn = dir ? (t - 1) : (t + 1);
            __builtin_prefetch(xp + (((size_t)tn * BTILES + bx) * 1024 +
                                     dir * 512 + 16 * wave) * 16);
        }

        // accumulators initialized from precomputed x-projection (+biases)
        v8f acc[4];
#pragma unroll
        for (int g = 0; g < 4; ++g) {
            int gcol = 128 * g + 16 * wave + col;
            v8h cv = *(const v8h*)(xp + (ctile + gcol) * 16 + rowb);
#pragma unroll
            for (int r = 0; r < 8; ++r) acc[g][r] = (float)cv[r];
        }

        // gates += h @ W_hh^T  (A from LDS, B resident in VGPRs)
#pragma unroll
        for (int kc = 0; kc < 4; ++kc) {
            int row = lane & 15, hi = (lane >> 4) * 8;
            const f16* p = &h_lds[row * 128 + kc * 32 + hi];
            v8h lo = *(const v8h*)p;
            v8h hv = *(const v8h*)(p + 16);
            v16h va;
#pragma unroll
            for (int i = 0; i < 8; ++i) { va[i] = lo[i]; va[8 + i] = hv[i]; }
#pragma unroll
            for (int g = 0; g < 4; ++g)
                acc[g] = wmma_f16(va, bf[g][kc], acc[g]);
        }
        __syncthreads();   // all A reads of h_lds done before overwrite

#pragma unroll
        for (int r = 0; r < 8; ++r) {
            float iv = sigm_f(acc[0][r]);
            float fv = sigm_f(acc[1][r]);
            float gv = tanh_f(acc[2][r]);
            float ov = sigm_f(acc[3][r]);
            float cc = fv * cst[r] + iv * gv;
            cst[r] = cc;
            float hv = ov * tanh_f(cc);

            h_lds[(rowb + r) * 128 + 16 * wave + col] = (f16)hv;
            if (!layer2) {
                size_t m = (size_t)t * B_N + (b0 + rowb + r);   // time-major
                h_out[m * 256 + dir * 128 + 16 * wave + col] = (f16)hv;
            } else {
                if (t < 5) pm0[r] = fmaxf(pm0[r], hv);
                else       pm1[r] = fmaxf(pm1[r], hv);
            }
        }
        __syncthreads();   // new h visible to all waves
    }

    if (layer2) {
        int c2 = dir * 128 + 16 * wave + col;     // concat channel 0..255
#pragma unroll
        for (int r = 0; r < 8; ++r) {
            size_t b = (size_t)(b0 + rowb + r);
            pooled[b * 512 + 2 * c2 + 0] = (f16)pm0[r];   // feature = 2c+pool
            pooled[b * 512 + 2 * c2 + 1] = (f16)pm1[r];
        }
    }
}

// ------------------------------------------------------------- FC head ------
// grid: B/16, block 32 (1 wave). fc1 via WMMA (K=512, N=64), relu, fc2 dot.
__global__ __launch_bounds__(32)
void fc_head(const f16* __restrict__ pooled, const f16* __restrict__ w1,
             const float* __restrict__ b1, const float* __restrict__ w2,
             const float* __restrict__ b2, float* __restrict__ out) {
    __shared__ float hbuf[16 * 64];
    int lane = threadIdx.x & 31;
    int m0 = blockIdx.x * 16;

    v8f acc[4] = {v8f{}, v8f{}, v8f{}, v8f{}};
    for (int kc = 0; kc < 16; ++kc) {
        int kb = kc * 32;
        v16h va = load_a_frag(pooled, 512, m0, kb, lane);
#pragma unroll
        for (int nt = 0; nt < 4; ++nt) {
            v16h vb = load_b_frag(w1, 512, nt * 16, kb, lane);
            acc[nt] = wmma_f16(va, vb, acc[nt]);
        }
    }
    int col = lane & 15, rowb = (lane >> 4) * 8;
#pragma unroll
    for (int nt = 0; nt < 4; ++nt) {
        int n = nt * 16 + col;
        float bb = b1[n];
#pragma unroll
        for (int r = 0; r < 8; ++r)
            hbuf[(rowb + r) * 64 + n] = fmaxf(acc[nt][r] + bb, 0.0f);
    }
    __syncthreads();
    if (lane < 16) {
        float s = b2[0];
#pragma unroll
        for (int j = 0; j < 64; ++j) s += hbuf[lane * 64 + j] * w2[j];
        out[m0 + lane] = s;
    }
}

// --------------------------------------------------------------- launcher ---

extern "C" void kernel_launch(void* const* d_in, const int* in_sizes, int n_in,
                              void* d_out, int out_size, void* d_ws, size_t ws_size,
                              hipStream_t stream) {
    (void)in_sizes; (void)n_in; (void)out_size; (void)ws_size;
    const float* x       = (const float*)d_in[0];
    const float* w_ih1_f = (const float*)d_in[1];
    const float* w_hh1_f = (const float*)d_in[2];
    const float* b_ih1_f = (const float*)d_in[3];
    const float* b_hh1_f = (const float*)d_in[4];
    const float* w_ih1_b = (const float*)d_in[5];
    const float* w_hh1_b = (const float*)d_in[6];
    const float* b_ih1_b = (const float*)d_in[7];
    const float* b_hh1_b = (const float*)d_in[8];
    const float* w_ih2_f = (const float*)d_in[9];
    const float* w_hh2_f = (const float*)d_in[10];
    const float* b_ih2_f = (const float*)d_in[11];
    const float* b_hh2_f = (const float*)d_in[12];
    const float* w_ih2_b = (const float*)d_in[13];
    const float* w_hh2_b = (const float*)d_in[14];
    const float* b_ih2_b = (const float*)d_in[15];
    const float* b_hh2_b = (const float*)d_in[16];
    const float* fc1_w   = (const float*)d_in[17];
    const float* fc1_b   = (const float*)d_in[18];
    const float* fc2_w   = (const float*)d_in[19];
    const float* fc2_b   = (const float*)d_in[20];
    float* outp = (float*)d_out;

    char* ws = (char*)d_ws;
    size_t off = 0;
    auto carve = [&](size_t bytes) -> char* {
        char* p = ws + off;
        off = (off + bytes + 255) & ~(size_t)255;
        return p;
    };
    f16* x16      = (f16*)carve((size_t)BT * K1PAD * 2);    // 60 MB (time-major)
    f16* h1_16    = (f16*)carve((size_t)BT * 256 * 2);      // 80 MB (time-major)
    f16* xp       = (f16*)carve((size_t)BT * 1024 * 2);     // 320 MB (reused L1/L2)
    f16* pooled16 = (f16*)carve((size_t)B_N * 512 * 2);     // 16 MB
    f16* wih1f16  = (f16*)carve((size_t)G_N * K1PAD * 2);
    f16* wih1b16  = (f16*)carve((size_t)G_N * K1PAD * 2);
    f16* whh1f16  = (f16*)carve((size_t)G_N * H_N * 2);
    f16* whh1b16  = (f16*)carve((size_t)G_N * H_N * 2);
    f16* wih2f16  = (f16*)carve((size_t)G_N * K2PAD * 2);
    f16* wih2b16  = (f16*)carve((size_t)G_N * K2PAD * 2);
    f16* whh2f16  = (f16*)carve((size_t)G_N * H_N * 2);
    f16* whh2b16  = (f16*)carve((size_t)G_N * H_N * 2);
    f16* fc1w16   = (f16*)carve((size_t)64 * 512 * 2);

    auto cvt = [&](const float* s, f16* d, int rows, int ks, int kd) {
        size_t tot = (size_t)rows * kd;
        cvt_pad_f16<<<dim3((unsigned)((tot + 255) / 256)), dim3(256), 0, stream>>>(
            s, d, rows, ks, kd);
    };
    {
        size_t tot = (size_t)BT * K1PAD;
        cvt_x_tmajor<<<dim3((unsigned)((tot + 255) / 256)), dim3(256), 0, stream>>>(
            x, x16);
    }
    cvt(w_ih1_f, wih1f16, G_N, I_N,  K1PAD);
    cvt(w_ih1_b, wih1b16, G_N, I_N,  K1PAD);
    cvt(w_hh1_f, whh1f16, G_N, H_N,  H_N);
    cvt(w_hh1_b, whh1b16, G_N, H_N,  H_N);
    cvt(w_ih2_f, wih2f16, G_N, K2PAD, K2PAD);
    cvt(w_ih2_b, wih2b16, G_N, K2PAD, K2PAD);
    cvt(w_hh2_f, whh2f16, G_N, H_N,  H_N);
    cvt(w_hh2_b, whh2b16, G_N, H_N,  H_N);
    cvt(fc1_w,   fc1w16,  64,  512,  512);

    // ---- layer 1 ----
    xproj_gemm<<<dim3(BT / 16 / 16, 8, 2), dim3(256), 0, stream>>>(
        x16, K1PAD, K1PAD / 32, wih1f16, wih1b16,
        b_ih1_f, b_hh1_f, b_ih1_b, b_hh1_b, xp);
    lstm_layer<<<dim3(BTILES, 2), dim3(256), 0, stream>>>(
        xp, whh1f16, whh1b16, h1_16, nullptr, 0);

    // ---- layer 2 ----
    xproj_gemm<<<dim3(BT / 16 / 16, 8, 2), dim3(256), 0, stream>>>(
        h1_16, K2PAD, K2PAD / 32, wih2f16, wih2b16,
        b_ih2_f, b_hh2_f, b_ih2_b, b_hh2_b, xp);
    lstm_layer<<<dim3(BTILES, 2), dim3(256), 0, stream>>>(
        xp, whh2f16, whh2b16, nullptr, pooled16, 1);

    // ---- pooled -> FC head ----
    fc_head<<<dim3(B_N / 16), dim3(32), 0, stream>>>(
        pooled16, fc1w16, fc1_b, fc2_w, fc2_b, outp);
}
